// Model_59648505807222
// MI455X (gfx1250) — matrix operand order repaired and checked by reading, not compiled, not run on hardware
//
#include <hip/hip_runtime.h>
#include <hip/hip_bf16.h>

// ---------------- model constants ----------------
#define Bn    32
#define Lseq  512      // LS == LD == 512
#define Dm    512
#define Hn    8
#define En    64
#define Mn    64       // top modes
#define Fn    257      // rfft bins of 512
#define DFFn  2048
#define CINn  21
#define COUTn 21
#define PREDn 256
#define LABELn 256
#define BHE   (Bn*Hn*En)       // 16384
#define BHEM  (Bn*Hn*En*Mn)    // 1048576
#define BLD   (Bn*Lseq*Dm)     // 8388608

typedef __attribute__((ext_vector_type(16))) _Float16 v16h;
typedef __attribute__((ext_vector_type(8)))  float    v8f;

__device__ __forceinline__ float gelu_tanh(float x) {
  float x3 = x * x * x;
  return 0.5f * x * (1.f + tanhf(0.7978845608028654f * (x + 0.044715f * x3)));
}

// ============ WMMA GEMM: C[M,N] = A[M,K] @ B[K,N] (+bias)(+resid)(gelu) ============
// f32 in/out, f16 compute, f32 accumulate.
// Block tile 128x64, 4 waves; wave tile 32x64 = 2x4 WMMA(16x16x32) per k-step.
// LDS staged in *fragment order* so each lane fragment = 2x ds_load_b128.
#define TM 128
#define TN 64
#define APAD 48   // halves per row (96 B, 32B-aligned v16h slices)

__global__ __launch_bounds__(128) void gemm_wmma_kernel(
    const float* __restrict__ A, const float* __restrict__ Bm,
    const float* __restrict__ bias, const float* __restrict__ resid,
    float* __restrict__ C, int Mr, int N, int K, int act) {
  __shared__ _Float16 As[TM][APAD];  // As[r][k]
  __shared__ _Float16 Bs[TN][APAD];  // Bs[col][k] (transposed tile)
  const int tid  = threadIdx.x;
  const int wave = tid >> 5, lane = tid & 31;
  const int row0 = blockIdx.y * TM, col0 = blockIdx.x * TN;
  const int mloc = lane & 15, kh = lane >> 4;

  v8f acc[2][4];
#pragma unroll
  for (int mt = 0; mt < 2; ++mt)
#pragma unroll
    for (int nt = 0; nt < 4; ++nt)
#pragma unroll
      for (int q = 0; q < 8; ++q) acc[mt][nt][q] = 0.f;

#pragma unroll 2
  for (int k0 = 0; k0 < K; k0 += 32) {
    // ---- stage A tile (128x32) into fragment-ordered LDS ----
    for (int i = tid; i < TM * 32; i += 128) {
      int r = i >> 5, c = i & 31;
      int pos = c;
      As[r][pos] = (_Float16)A[(size_t)(row0 + r) * K + (k0 + c)];
    }
    // ---- stage B tile (32x64) transposed: Bs[c][r] ----
    for (int i = tid; i < 32 * TN; i += 128) {
      int r = i >> 6, c = i & 63;
      Bs[c][r] = (_Float16)Bm[(size_t)(k0 + r) * N + (col0 + c)];
    }
    if (k0 + 32 < K) {  // hint next tiles (global_prefetch_b8)
      __builtin_prefetch(&A[(size_t)(row0 + tid) * K + k0 + 32], 0, 1);
      __builtin_prefetch(&Bm[(size_t)(k0 + 32 + (tid >> 2)) * N + col0], 0, 1);
    }
    __syncthreads();

    v16h afrag[2];
#pragma unroll
    for (int mt = 0; mt < 2; ++mt)
      afrag[mt] = *(const v16h*)&As[wave * 32 + mt * 16 + mloc][kh * 16];
#pragma unroll
    for (int nt = 0; nt < 4; ++nt) {
      v16h bfrag = *(const v16h*)&Bs[nt * 16 + mloc][kh * 16];
      acc[0][nt] = __builtin_amdgcn_wmma_f32_16x16x32_f16(
          false, afrag[0], false, bfrag, (short)0, acc[0][nt], false, false);
      acc[1][nt] = __builtin_amdgcn_wmma_f32_16x16x32_f16(
          false, afrag[1], false, bfrag, (short)0, acc[1][nt], false, false);
    }
    __syncthreads();
  }

  // epilogue: C/D layout — VGPR v holds M = v + 8*(lane>>4), N = lane&15
#pragma unroll
  for (int mt = 0; mt < 2; ++mt) {
#pragma unroll
    for (int nt = 0; nt < 4; ++nt) {
#pragma unroll
      for (int v = 0; v < 8; ++v) {
        int grow = row0 + wave * 32 + mt * 16 + v + 8 * kh;
        int gcol = col0 + nt * 16 + mloc;
        float val = acc[mt][nt][v];
        if (bias)  val += bias[gcol];
        if (resid) val += resid[(size_t)grow * N + gcol];
        if (act == 1) val = gelu_tanh(val);
        C[(size_t)grow * N + gcol] = val;
      }
    }
  }
}

// ============ series_decomp: moving mean (K=25, edge pad) ============
__global__ void decomp_kernel(const float* __restrict__ x, float* __restrict__ sout,
                              float* __restrict__ mout, int L, int C, int accum, int total) {
  int i = blockIdx.x * 256 + threadIdx.x;
  if (i >= total) return;
  int c = i % C; int bl = i / C; int l = bl % L; int b = bl / L;
  float s = 0.f;
#pragma unroll 5
  for (int j = l - 12; j <= l + 12; ++j) {
    int jj = j < 0 ? 0 : (j > L - 1 ? L - 1 : j);
    s += x[((size_t)b * L + jj) * C + c];
  }
  float m = s * (1.f / 25.f);
  sout[i] = x[i] - m;
  if (accum) mout[i] += m; else mout[i] = m;
}

// mean over time of x_enc -> meanc[B,CIN]
__global__ void meanc_kernel(const float* __restrict__ x, float* __restrict__ meanc) {
  int i = blockIdx.x * 256 + threadIdx.x;
  if (i >= Bn * CINn) return;
  int c = i % CINn, b = i / CINn;
  float s = 0.f;
  for (int l = 0; l < Lseq; ++l) s += x[((size_t)b * Lseq + l) * CINn + c];
  meanc[i] = s * (1.f / (float)Lseq);
}

// build low_init / high_init [B,512,CIN]
__global__ void lowhigh_kernel(const float* __restrict__ s, const float* __restrict__ m,
                               const float* __restrict__ meanc,
                               float* __restrict__ low, float* __restrict__ high) {
  int i = blockIdx.x * 256 + threadIdx.x;
  if (i >= Bn * Lseq * CINn) return;
  int c = i % CINn; int bl = i / CINn; int l = bl % Lseq; int b = bl / Lseq;
  if (l < LABELn) {
    size_t src = ((size_t)b * Lseq + (Lseq - LABELn + l)) * CINn + c;
    low[i] = m[src]; high[i] = s[src];
  } else {
    low[i] = meanc[b * CINn + c]; high[i] = 0.f;
  }
}

// DataEmbedding_wo_pos: circular conv(k=3, CIN->D) + mark @ tw
__global__ void embed_kernel(const float* __restrict__ x, const float* __restrict__ mark,
                             const float* __restrict__ ck, const float* __restrict__ tw,
                             float* __restrict__ out) {
  int i = blockIdx.x * 256 + threadIdx.x;
  if (i >= Bn * Lseq * Dm) return;
  int d = i % Dm; int bl = i / Dm; int l = bl % Lseq; int b = bl / Lseq;
  float acc = 0.f;
  for (int t = 0; t < 3; ++t) {
    int ll = (l - 1 + t + Lseq) % Lseq;
    const float* xp = x + ((size_t)b * Lseq + ll) * CINn;
    const float* kp = ck + ((size_t)t * CINn) * Dm + d;
    for (int c = 0; c < CINn; ++c) acc += xp[c] * kp[(size_t)c * Dm];
  }
  const float* mp = mark + ((size_t)b * Lseq + l) * 4;
#pragma unroll
  for (int j = 0; j < 4; ++j) acc += mp[j] * tw[(size_t)j * Dm + d];
  out[i] = acc;
}

// ============ 512-pt FFT kernels ============
// forward rfft of x[B,L,D] viewed as [B,H,E,L] -> fre/fim [BHE, 257]
__global__ __launch_bounds__(256) void rfft512_kernel(const float* __restrict__ x,
                                                      float* __restrict__ fre,
                                                      float* __restrict__ fim) {
  const int n = blockIdx.x, tid = threadIdx.x;
  const int b = n / (Hn * En), he = n % (Hn * En);
  __shared__ float sre[512], sim[512];
  for (int l = tid; l < 512; l += 256) {
    int r = __brev((unsigned)l) >> 23;
    sre[r] = x[((size_t)b * Lseq + l) * Dm + he];
    sim[r] = 0.f;
  }
  for (int s = 1; s <= 9; ++s) {
    __syncthreads();
    int m = 1 << s, half = m >> 1;
    int j = tid & (half - 1);
    int k = ((tid >> (s - 1)) << s) + j;
    float ang = -6.283185307179586f * (float)j / (float)m;
    float wr = __cosf(ang), wi = __sinf(ang);
    float vr = sre[k + half], vi = sim[k + half];
    float tr = vr * wr - vi * wi, ti = vr * wi + vi * wr;
    float ur = sre[k], ui = sim[k];
    sre[k] = ur + tr;        sim[k] = ui + ti;
    sre[k + half] = ur - tr; sim[k + half] = ui - ti;
  }
  __syncthreads();
  for (int f = tid; f < Fn; f += 256) {
    fre[(size_t)n * Fn + f] = sre[f];
    fim[(size_t)n * Fn + f] = sim[f];
  }
}

// amplitude mean over (B,H,E)
__global__ __launch_bounds__(256) void amp_kernel(const float* __restrict__ fre,
                                                  const float* __restrict__ fim,
                                                  float* __restrict__ amp) {
  const int f = blockIdx.x, tid = threadIdx.x;
  __shared__ float red[256];
  float s = 0.f;
  for (int n = tid; n < BHE; n += 256) {
    float re = fre[(size_t)n * Fn + f], im = fim[(size_t)n * Fn + f];
    s += sqrtf(re * re + im * im);
  }
  red[tid] = s; __syncthreads();
  for (int o = 128; o > 0; o >>= 1) { if (tid < o) red[tid] += red[tid + o]; __syncthreads(); }
  if (tid == 0) amp[f] = red[0] * (1.f / (float)BHE);
}

// top-64 (descending, first-index ties) of amp[257]
__global__ void topk_kernel(const float* __restrict__ amp, int* __restrict__ idx) {
  __shared__ float a[Fn];
  int tid = threadIdx.x;
  for (int f = tid; f < Fn; f += blockDim.x) a[f] = amp[f];
  __syncthreads();
  if (tid == 0) {
    for (int m = 0; m < Mn; ++m) {
      int best = 0; float bv = a[0];
      for (int f = 1; f < Fn; ++f) if (a[f] > bv) { bv = a[f]; best = f; }
      idx[m] = best; a[best] = -1e30f;
    }
  }
}

// gather selected modes: xs[BHE,M]
__global__ void gather_kernel(const float* __restrict__ fre, const float* __restrict__ fim,
                              const int* __restrict__ idx,
                              float* __restrict__ xre, float* __restrict__ xim) {
  int i = blockIdx.x * 256 + threadIdx.x;
  if (i >= BHEM) return;
  int m = i % Mn; int n = i / Mn; int f = idx[m];
  xre[i] = fre[(size_t)n * Fn + f];
  xim[i] = fim[(size_t)n * Fn + f];
}

// outs[b,h,o,m] = sum_e xs[b,h,e,m] * (wre+1j*wim)[h,e,o,m]
__global__ void einsum_cw_kernel(const float* __restrict__ xre, const float* __restrict__ xim,
                                 const float* __restrict__ wre, const float* __restrict__ wim,
                                 float* __restrict__ ore, float* __restrict__ oim) {
  int i = blockIdx.x * 256 + threadIdx.x;
  if (i >= BHEM) return;
  int m = i % Mn; int o = (i / Mn) % En; int h = (i / (Mn * En)) % Hn; int b = i / (Mn * En * Hn);
  size_t xb = ((size_t)(b * Hn + h) * En) * Mn + m;           // +e*Mn
  size_t wb = (size_t)h * En * En * Mn + (size_t)o * Mn + m;  // +e*En*Mn
  float ar = 0.f, ai = 0.f;
  for (int e = 0; e < En; ++e) {
    float xr = xre[xb + (size_t)e * Mn], xi = xim[xb + (size_t)e * Mn];
    float wr = wre[wb + (size_t)e * En * Mn], wi = wim[wb + (size_t)e * En * Mn];
    ar += xr * wr - xi * wi;
    ai += xr * wi + xi * wr;
  }
  ore[i] = ar; oim[i] = ai;
}

__device__ __forceinline__ void ctanh_dev(float a, float b, float& tre, float& tim) {
  float ta = 2.f * a, tb = 2.f * b;
  if (fabsf(ta) > 30.f) { tre = ta > 0.f ? 1.f : -1.f; tim = 0.f; return; }
  float d = coshf(ta) + __cosf(tb);
  tre = sinhf(ta) / d; tim = __sinf(tb) / d;
}

// qk[b,h,x,y] = ctanh( sum_e qs[b,h,e,x] * ks[b,h,e,y] )
__global__ void qk_tanh_kernel(const float* __restrict__ qre, const float* __restrict__ qim,
                               const float* __restrict__ kre, const float* __restrict__ kim,
                               float* __restrict__ ore, float* __restrict__ oim) {
  int i = blockIdx.x * 256 + threadIdx.x;
  if (i >= Bn * Hn * Mn * Mn) return;
  int y = i % Mn; int x = (i / Mn) % Mn; int h = (i / (Mn * Mn)) % Hn; int b = i / (Mn * Mn * Hn);
  size_t qb = ((size_t)(b * Hn + h) * En) * Mn + x;
  size_t kb = ((size_t)(b * Hn + h) * En) * Mn + y;
  float ar = 0.f, ai = 0.f;
  for (int e = 0; e < En; ++e) {
    float a0 = qre[qb + (size_t)e * Mn], a1 = qim[qb + (size_t)e * Mn];
    float b0 = kre[kb + (size_t)e * Mn], b1 = kim[kb + (size_t)e * Mn];
    ar += a0 * b0 - a1 * b1;
    ai += a0 * b1 + a1 * b0;
  }
  float tr, ti; ctanh_dev(ar, ai, tr, ti);
  ore[i] = tr; oim[i] = ti;
}

// qkv[b,h,e,x] = sum_y qk[b,h,x,y] * ks[b,h,e,y]
__global__ void qkv_kernel(const float* __restrict__ qkre, const float* __restrict__ qkim,
                           const float* __restrict__ kre, const float* __restrict__ kim,
                           float* __restrict__ ore, float* __restrict__ oim) {
  int i = blockIdx.x * 256 + threadIdx.x;
  if (i >= BHEM) return;
  int x = i % Mn; int e = (i / Mn) % En; int h = (i / (Mn * En)) % Hn; int b = i / (Mn * En * Hn);
  size_t qb = ((size_t)(b * Hn + h) * Mn + x) * Mn;
  size_t kb = ((size_t)(b * Hn + h) * En + e) * Mn;
  float ar = 0.f, ai = 0.f;
  for (int y = 0; y < Mn; ++y) {
    float a0 = qkre[qb + y], a1 = qkim[qb + y];
    float b0 = kre[kb + y],  b1 = kim[kb + y];
    ar += a0 * b0 - a1 * b1;
    ai += a0 * b1 + a1 * b0;
  }
  ore[i] = ar; oim[i] = ai;
}

// scatter modes into spectrum, hermitian extend, inverse 512-pt FFT, real part * scale
__global__ __launch_bounds__(256) void irfft512_scatter_kernel(
    const float* __restrict__ ore, const float* __restrict__ oim,
    const int* __restrict__ idx, float* __restrict__ y, float scale) {
  const int n = blockIdx.x, tid = threadIdx.x;
  __shared__ float sre[512], sim[512];
  for (int l = tid; l < 512; l += 256) { sre[l] = 0.f; sim[l] = 0.f; }
  __syncthreads();
  if (tid < Mn) {
    int f = idx[tid];
    sre[f] = ore[(size_t)n * Mn + tid];
    sim[f] = oim[(size_t)n * Mn + tid];
  }
  __syncthreads();
  if (tid >= 1 && tid <= 255) { sre[512 - tid] = sre[tid]; sim[512 - tid] = -sim[tid]; }
  __syncthreads();
  for (int i = tid; i < 512; i += 256) {
    int r = __brev((unsigned)i) >> 23;
    if (i < r) {
      float t0 = sre[i]; sre[i] = sre[r]; sre[r] = t0;
      float t1 = sim[i]; sim[i] = sim[r]; sim[r] = t1;
    }
  }
  for (int s = 1; s <= 9; ++s) {
    __syncthreads();
    int m = 1 << s, half = m >> 1;
    int j = tid & (half - 1);
    int k = ((tid >> (s - 1)) << s) + j;
    float ang = 6.283185307179586f * (float)j / (float)m;   // inverse: +i
    float wr = __cosf(ang), wi = __sinf(ang);
    float vr = sre[k + half], vi = sim[k + half];
    float tr = vr * wr - vi * wi, ti = vr * wi + vi * wr;
    float ur = sre[k], ui = sim[k];
    sre[k] = ur + tr;        sim[k] = ui + ti;
    sre[k + half] = ur - tr; sim[k + half] = ui - ti;
  }
  __syncthreads();
  const int b = n / (Hn * En), h = (n / En) % Hn, o = n % En;
  const float sc = scale * (1.f / 512.f);
  for (int l = tid; l < 512; l += 256)
    y[((size_t)b * Lseq + l) * Dm + h * En + o] = sre[l] * sc;
}

// ============ my_Layernorm ============
__global__ __launch_bounds__(256) void ln_row_kernel(const float* __restrict__ x,
                                                     const float* __restrict__ g,
                                                     const float* __restrict__ bb,
                                                     float* __restrict__ out) {
  const int bl = blockIdx.x, tid = threadIdx.x;
  __shared__ float r1[256], r2[256];
  const float* xp = x + (size_t)bl * Dm;
  float s = 0.f, s2 = 0.f;
  for (int d = tid; d < Dm; d += 256) { float v = xp[d]; s += v; s2 += v * v; }
  r1[tid] = s; r2[tid] = s2; __syncthreads();
  for (int o = 128; o > 0; o >>= 1) {
    if (tid < o) { r1[tid] += r1[tid + o]; r2[tid] += r2[tid + o]; }
    __syncthreads();
  }
  float mu = r1[0] * (1.f / (float)Dm);
  float var = r2[0] * (1.f / (float)Dm) - mu * mu;
  float rs = rsqrtf(var + 1e-5f);
  for (int d = tid; d < Dm; d += 256)
    out[(size_t)bl * Dm + d] = (xp[d] - mu) * rs * g[d] + bb[d];
}

__global__ void colmean_kernel(const float* __restrict__ x, float* __restrict__ cm) {
  int i = blockIdx.x * 256 + threadIdx.x;
  if (i >= Bn * Dm) return;
  int d = i % Dm, b = i / Dm;
  float s = 0.f;
  for (int l = 0; l < Lseq; ++l) s += x[((size_t)b * Lseq + l) * Dm + d];
  cm[i] = s * (1.f / (float)Lseq);
}

__global__ void submean_kernel(float* __restrict__ x, const float* __restrict__ cm) {
  int i = blockIdx.x * 256 + threadIdx.x;
  if (i >= BLD) return;
  int d = i % Dm; int b = i / (Lseq * Dm);
  x[i] -= cm[(size_t)b * Dm + d];
}

// low_part = low_init + circ_conv(trend, dec_tp)
__global__ void trendconv_kernel(const float* __restrict__ trend, const float* __restrict__ tp,
                                 const float* __restrict__ lowinit, float* __restrict__ lowpart) {
  int i = blockIdx.x * 256 + threadIdx.x;
  if (i >= Bn * Lseq * COUTn) return;
  int o = i % COUTn; int bl = i / COUTn; int l = bl % Lseq; int b = bl / Lseq;
  float acc = lowinit[i];
  for (int t = 0; t < 3; ++t) {
    int ll = (l - 1 + t + Lseq) % Lseq;
    const float* xp = trend + ((size_t)b * Lseq + ll) * Dm;
    const float* kp = tp + (size_t)t * Dm * COUTn + o;
    for (int c = 0; c < Dm; ++c) acc += xp[c] * kp[(size_t)c * COUTn];
  }
  lowpart[i] = acc;
}

// high_part = ln @ proj_W + proj_b
__global__ void proj_kernel(const float* __restrict__ x, const float* __restrict__ W,
                            const float* __restrict__ bvec, float* __restrict__ out) {
  int i = blockIdx.x * 256 + threadIdx.x;
  if (i >= Bn * Lseq * COUTn) return;
  int o = i % COUTn; int bl = i / COUTn;
  const float* xp = x + (size_t)bl * Dm;
  float acc = bvec[o];
  for (int d = 0; d < Dm; ++d) acc += xp[d] * W[(size_t)d * COUTn + o];
  out[i] = acc;
}

__global__ void final_kernel(const float* __restrict__ lowpart,
                             const float* __restrict__ highpart, float* __restrict__ out) {
  int i = blockIdx.x * 256 + threadIdx.x;
  if (i >= Bn * PREDn * COUTn) return;
  int o = i % COUTn; int p = (i / COUTn) % PREDn; int b = i / (COUTn * PREDn);
  size_t src = ((size_t)b * Lseq + (Lseq - PREDn) + p) * COUTn + o;
  out[i] = lowpart[src] + highpart[src];
}

// =====================================================================
extern "C" void kernel_launch(void* const* d_in, const int* in_sizes, int n_in,
                              void* d_out, int out_size, void* d_ws, size_t ws_size,
                              hipStream_t stream) {
  (void)in_sizes; (void)n_in; (void)out_size; (void)ws_size;
  const float* x_enc      = (const float*)d_in[0];
  const float* x_mark_enc = (const float*)d_in[1];
  const float* x_mark_dec = (const float*)d_in[3];
  const float* emb_enc_k  = (const float*)d_in[4];
  const float* emb_enc_t  = (const float*)d_in[5];
  const float* emb_dec_k  = (const float*)d_in[6];
  const float* emb_dec_t  = (const float*)d_in[7];
  const float* fw_enc_re  = (const float*)d_in[8];
  const float* fw_enc_im  = (const float*)d_in[9];
  const float* fw_dec_re  = (const float*)d_in[10];
  const float* fw_dec_im  = (const float*)d_in[11];
  const float* fw_cr_re   = (const float*)d_in[12];
  const float* fw_cr_im   = (const float*)d_in[13];
  const float* enc_Wq = (const float*)d_in[14]; const float* enc_bq = (const float*)d_in[15];
  const float* enc_Wo = (const float*)d_in[16]; const float* enc_bo = (const float*)d_in[17];
  const float* enc_c1 = (const float*)d_in[18]; const float* enc_c2 = (const float*)d_in[19];
  const float* enc_ng = (const float*)d_in[20]; const float* enc_nb = (const float*)d_in[21];
  const float* ds_Wq  = (const float*)d_in[22]; const float* ds_bq  = (const float*)d_in[23];
  const float* ds_Wo  = (const float*)d_in[24]; const float* ds_bo  = (const float*)d_in[25];
  const float* dc_Wq  = (const float*)d_in[26]; const float* dc_bq  = (const float*)d_in[27];
  const float* dc_Wk  = (const float*)d_in[28]; const float* dc_bk  = (const float*)d_in[29];
  const float* dc_Wo  = (const float*)d_in[30]; const float* dc_bo  = (const float*)d_in[31];
  const float* dec_c1 = (const float*)d_in[32]; const float* dec_c2 = (const float*)d_in[33];
  const float* dec_tp = (const float*)d_in[34];
  const float* dec_ng = (const float*)d_in[35]; const float* dec_nb = (const float*)d_in[36];
  const float* proj_W = (const float*)d_in[37]; const float* proj_b = (const float*)d_in[38];
  float* out = (float*)d_out;

  // ---- bump allocator over d_ws ----
  size_t cur = 0;
  auto alloc = [&](size_t nfl) -> float* {
    float* p = (float*)((char*)d_ws + cur);
    cur = (cur + nfl * sizeof(float) + 255) & ~(size_t)255;
    return p;
  };
  const size_t BLC = (size_t)Bn * Lseq * CINn;
  float* f_s_enc  = alloc(BLC);
  float* f_m_enc  = alloc(BLC);
  float* meanc    = alloc(Bn * CINn);
  float* low_init = alloc(BLC);
  float* high_init= alloc(BLC);
  float* f_enc  = alloc(BLD);
  float* f_q    = alloc(BLD);
  float* f_y    = alloc(BLD);
  float* f_a    = alloc(BLD);
  float* f_x1   = alloc(BLD);
  float* f_x2   = alloc(BLD);
  float* f_x3   = alloc(BLD);
  float* f_dec  = alloc(BLD);
  float* f_k    = alloc(BLD);
  float* f_tr   = alloc(BLD);
  float* f_dump = alloc(BLD);
  float* f_ln   = alloc(BLD);
  float* f_ffn  = alloc((size_t)Bn * Lseq * DFFn);
  float* xf_re = alloc((size_t)BHE * Fn); float* xf_im = alloc((size_t)BHE * Fn);
  float* kf_re = alloc((size_t)BHE * Fn); float* kf_im = alloc((size_t)BHE * Fn);
  float* xs_re = alloc(BHEM); float* xs_im = alloc(BHEM);
  float* ks_re = alloc(BHEM); float* ks_im = alloc(BHEM);
  float* os_re = alloc(BHEM); float* os_im = alloc(BHEM);
  float* qk_re = alloc(BHEM); float* qk_im = alloc(BHEM);
  float* qv_re = alloc(BHEM); float* qv_im = alloc(BHEM);
  float* amp1  = alloc(Fn);   float* amp2  = alloc(Fn);
  int*   idx1  = (int*)alloc(64);
  int*   idx2  = (int*)alloc(64);
  float* cmean = alloc(Bn * Dm);
  float* lowp  = alloc(BLC);
  float* highp = alloc(BLC);

  auto g1 = [](size_t n) { return dim3((unsigned)((n + 255) / 256)); };
  auto gemm = [&](const float* A, const float* Bm, const float* bias, const float* resid,
                  float* C, int Mr, int N, int K, int act) {
    dim3 grd(N / TN, Mr / TM);
    gemm_wmma_kernel<<<grd, 128, 0, stream>>>(A, Bm, bias, resid, C, Mr, N, K, act);
  };
  const int BL = Bn * Lseq;  // 16384 rows in all big GEMMs

  auto freq_self = [&](const float* qbuf, const float* wre, const float* wim, float* ybuf) {
    rfft512_kernel<<<BHE, 256, 0, stream>>>(qbuf, xf_re, xf_im);
    amp_kernel<<<Fn, 256, 0, stream>>>(xf_re, xf_im, amp1);
    topk_kernel<<<1, 64, 0, stream>>>(amp1, idx1);
    gather_kernel<<<g1(BHEM), 256, 0, stream>>>(xf_re, xf_im, idx1, xs_re, xs_im);
    einsum_cw_kernel<<<g1(BHEM), 256, 0, stream>>>(xs_re, xs_im, wre, wim, os_re, os_im);
    irfft512_scatter_kernel<<<BHE, 256, 0, stream>>>(os_re, os_im, idx1, ybuf, 1.f);
  };

  // ---- input decomposition ----
  decomp_kernel<<<g1(BLC), 256, 0, stream>>>(x_enc, f_s_enc, f_m_enc, Lseq, CINn, 0, (int)BLC);
  meanc_kernel<<<g1(Bn * CINn), 256, 0, stream>>>(x_enc, meanc);
  lowhigh_kernel<<<g1(BLC), 256, 0, stream>>>(f_s_enc, f_m_enc, meanc, low_init, high_init);

  // ---- encoder ----
  embed_kernel<<<g1(BLD), 256, 0, stream>>>(x_enc, x_mark_enc, emb_enc_k, emb_enc_t, f_enc);
  for (int i = 0; i < 2; ++i) {
    const float* Wq = enc_Wq + (size_t)i * Dm * Dm;  const float* bq = enc_bq + (size_t)i * Dm;
    const float* Wo = enc_Wo + (size_t)i * Dm * Dm;  const float* bo = enc_bo + (size_t)i * Dm;
    const float* c1 = enc_c1 + (size_t)i * Dm * DFFn;
    const float* c2 = enc_c2 + (size_t)i * DFFn * Dm;
    gemm(f_enc, Wq, bq, nullptr, f_q, BL, Dm, Dm, 0);
    freq_self(f_q, fw_enc_re, fw_enc_im, f_y);
    gemm(f_y, Wo, bo, f_enc, f_a, BL, Dm, Dm, 0);
    decomp_kernel<<<g1(BLD), 256, 0, stream>>>(f_a, f_x1, f_dump, Lseq, Dm, 0, BLD);
    gemm(f_x1, c1, nullptr, nullptr, f_ffn, BL, DFFn, Dm, 1);
    gemm(f_ffn, c2, nullptr, f_x1, f_a, BL, Dm, DFFn, 0);
    decomp_kernel<<<g1(BLD), 256, 0, stream>>>(f_a, f_enc, f_dump, Lseq, Dm, 0, BLD);
  }
  ln_row_kernel<<<BL, 256, 0, stream>>>(f_enc, enc_ng, enc_nb, f_ln);
  colmean_kernel<<<g1(Bn * Dm), 256, 0, stream>>>(f_ln, cmean);
  submean_kernel<<<g1(BLD), 256, 0, stream>>>(f_ln, cmean);   // f_ln = final encoder output

  // ---- decoder ----
  embed_kernel<<<g1(BLD), 256, 0, stream>>>(high_init, x_mark_dec, emb_dec_k, emb_dec_t, f_dec);
  // self attention
  gemm(f_dec, ds_Wq, ds_bq, nullptr, f_q, BL, Dm, Dm, 0);
  freq_self(f_q, fw_dec_re, fw_dec_im, f_y);
  gemm(f_y, ds_Wo, ds_bo, f_dec, f_a, BL, Dm, Dm, 0);
  decomp_kernel<<<g1(BLD), 256, 0, stream>>>(f_a, f_x1, f_tr, Lseq, Dm, 0, BLD);   // t1
  // cross attention
  gemm(f_x1, dc_Wq, dc_bq, nullptr, f_q, BL, Dm, Dm, 0);
  gemm(f_ln, dc_Wk, dc_bk, nullptr, f_k, BL, Dm, Dm, 0);
  rfft512_kernel<<<BHE, 256, 0, stream>>>(f_q, xf_re, xf_im);
  rfft512_kernel<<<BHE, 256, 0, stream>>>(f_k, kf_re, kf_im);
  amp_kernel<<<Fn, 256, 0, stream>>>(xf_re, xf_im, amp1);
  amp_kernel<<<Fn, 256, 0, stream>>>(kf_re, kf_im, amp2);
  topk_kernel<<<1, 64, 0, stream>>>(amp1, idx1);
  topk_kernel<<<1, 64, 0, stream>>>(amp2, idx2);
  gather_kernel<<<g1(BHEM), 256, 0, stream>>>(xf_re, xf_im, idx1, xs_re, xs_im);
  gather_kernel<<<g1(BHEM), 256, 0, stream>>>(kf_re, kf_im, idx2, ks_re, ks_im);
  qk_tanh_kernel<<<g1((size_t)Bn * Hn * Mn * Mn), 256, 0, stream>>>(xs_re, xs_im, ks_re, ks_im, qk_re, qk_im);
  qkv_kernel<<<g1(BHEM), 256, 0, stream>>>(qk_re, qk_im, ks_re, ks_im, qv_re, qv_im);
  einsum_cw_kernel<<<g1(BHEM), 256, 0, stream>>>(qv_re, qv_im, fw_cr_re, fw_cr_im, os_re, os_im);
  irfft512_scatter_kernel<<<BHE, 256, 0, stream>>>(os_re, os_im, idx1, f_y,
                                                   1.f / ((float)Dm * (float)Dm));
  gemm(f_y, dc_Wo, dc_bo, f_x1, f_a, BL, Dm, Dm, 0);
  decomp_kernel<<<g1(BLD), 256, 0, stream>>>(f_a, f_x2, f_tr, Lseq, Dm, 1, BLD);   // t2 (+=)
  // FFN
  gemm(f_x2, dec_c1, nullptr, nullptr, f_ffn, BL, DFFn, Dm, 1);
  gemm(f_ffn, dec_c2, nullptr, f_x2, f_a, BL, Dm, DFFn, 0);
  decomp_kernel<<<g1(BLD), 256, 0, stream>>>(f_a, f_x3, f_tr, Lseq, Dm, 1, BLD);   // t3 (+=)

  // trend branch
  trendconv_kernel<<<g1(BLC), 256, 0, stream>>>(f_tr, dec_tp, low_init, lowp);
  // seasonal branch
  ln_row_kernel<<<BL, 256, 0, stream>>>(f_x3, dec_ng, dec_nb, f_ln);
  colmean_kernel<<<g1(Bn * Dm), 256, 0, stream>>>(f_ln, cmean);
  submean_kernel<<<g1(BLD), 256, 0, stream>>>(f_ln, cmean);
  proj_kernel<<<g1(BLC), 256, 0, stream>>>(f_ln, proj_W, proj_b, highp);
  // output
  final_kernel<<<g1((size_t)Bn * PREDn * COUTn), 256, 0, stream>>>(lowp, highp, out);
}
